// MinkUNetBase_66185446031553
// MI455X (gfx1250) — matrix operand order repaired and checked by reading, not compiled, not run on hardware
//
#include <hip/hip_runtime.h>

typedef __attribute__((ext_vector_type(16))) _Float16 v16h;
typedef __attribute__((ext_vector_type(8)))  _Float16 h8;
typedef __attribute__((ext_vector_type(8)))  float    v8f;

#define LRELU(v) ((v) >= 0.f ? (v) : 0.1f*(v))

// ---------------------------------------------------------------------------
// Elementwise helpers
// ---------------------------------------------------------------------------

// out16[v*C+c] = f16( x * mask[v] ); src layout channels-first (NCDHW) or channels-last
__global__ void premul_kernel(const float* __restrict__ x, const float* __restrict__ m,
                              _Float16* __restrict__ out, int V, int C, int chfirst) {
  int i = blockIdx.x * blockDim.x + threadIdx.x;
  if (i >= V * C) return;
  int v = i / C, c = i - v * C;
  float xv = chfirst ? x[(size_t)c * V + v] : x[i];
  out[i] = (_Float16)(xv * m[v]);
}

// Pack conv weight [Ktot][Cout] (f32) into WMMA B-fragment order (f16):
// wp[(t*Cp + n)*32 + h] = W[32t+h][n], zero-padded in K (h beyond Ktot) and N (n>=Cout).
// With this layout a lane's whole B fragment is one contiguous 32B chunk.
__global__ void packB_kernel(const float* __restrict__ w, _Float16* __restrict__ wp,
                             int Ktot, int Cout, int Cp) {
  int i = blockIdx.x * blockDim.x + threadIdx.x;
  int Ktiles = (Ktot + 31) >> 5;
  if (i >= Ktiles * Cp * 32) return;
  int h = i & 31;
  int rest = i >> 5;
  int n = rest % Cp;
  int k = (rest / Cp) * 32 + h;
  float v = (k < Ktot && n < Cout) ? w[(size_t)k * Cout + n] : 0.f;
  wp[i] = (_Float16)v;
}

__global__ void concat_kernel(const float* __restrict__ a, const float* __restrict__ b,
                              float* __restrict__ out, int V, int C1, int C2) {
  int C = C1 + C2;
  int i = blockIdx.x * blockDim.x + threadIdx.x;
  if (i >= V * C) return;
  int v = i / C, c = i - v * C;
  out[i] = (c < C1) ? a[(size_t)v * C1 + c] : b[(size_t)v * C2 + (c - C1)];
}

__global__ void maskclip_kernel(const float* __restrict__ a, const float* __restrict__ b,
                                float* __restrict__ out, int V) {
  int i = blockIdx.x * blockDim.x + threadIdx.x;
  if (i < V) out[i] = fminf(a[i] + b[i], 1.f);
}

// d_out: y transposed to NCDHW [C][V], then mask [V]
__global__ void final_kernel(const float* __restrict__ y, const float* __restrict__ bias,
                             const float* __restrict__ m, float* __restrict__ out, int V, int C) {
  int i = blockIdx.x * blockDim.x + threadIdx.x;
  if (i >= V * C + V) return;
  if (i < V * C) {
    int v = i / C, c = i - v * C;
    out[(size_t)c * V + v] = (y[i] + bias[c]) * m[v];
  } else {
    int v = i - V * C;
    out[(size_t)C * V + v] = m[v];
  }
}

// ---------------------------------------------------------------------------
// Masked instance norm (per sample, per channel), fused variants.
// mode 0: y = inorm(x)*m ; mode 1: lrelu(inorm(x)*m) ; mode 2: lrelu(inorm(x)*m + res)
// grid.x = C, 256 threads. In-place (out==x) is safe.
// ---------------------------------------------------------------------------
__global__ void inorm_kernel(const float* __restrict__ x, const float* __restrict__ m,
                             const float* __restrict__ gamma, const float* __restrict__ beta,
                             const float* __restrict__ res, float* __restrict__ out,
                             int V, int C, int mode) {
  const int c = blockIdx.x;
  __shared__ float red[256];
  float s = 0.f, s2 = 0.f, sc = 0.f;
  for (int v = threadIdx.x; v < V; v += blockDim.x) {
    float mv = m[v];
    float xv = x[(size_t)v * C + c];
    s += xv * mv; s2 += xv * xv * mv; sc += mv;
  }
  float sums[3] = { s, s2, sc };
  float tot[3];
  for (int j = 0; j < 3; ++j) {
    red[threadIdx.x] = sums[j];
    __syncthreads();
    for (int o = 128; o > 0; o >>= 1) {
      if ((int)threadIdx.x < o) red[threadIdx.x] += red[threadIdx.x + o];
      __syncthreads();
    }
    tot[j] = red[0];
    __syncthreads();
  }
  float cnt  = fmaxf(tot[2], 1.f);
  float mean = tot[0] / cnt;
  float var  = fmaxf(tot[1] / cnt - mean * mean, 0.f);
  float inv  = rsqrtf(var + 1e-5f);
  float ga = gamma[c], be = beta[c];
  for (int v = threadIdx.x; v < V; v += blockDim.x) {
    float xv = x[(size_t)v * C + c];
    float y = ((xv - mean) * inv * ga + be) * m[v];
    if (mode == 2) y += res[(size_t)v * C + c];
    if (mode >= 1) y = LRELU(y);
    out[(size_t)v * C + c] = y;
  }
}

// ---------------------------------------------------------------------------
// Partial-conv implicit GEMM, one wave per 16x32 (voxels x Cout) tile:
// one A fragment feeds TWO v_wmma_f32_16x16x32_f16 issues against two
// pre-packed B fragments (halves A-gather cost per MAC).
// Fast path (Cin%32==0): per kernel-tap the voxel/bounds checks hoist out;
// A = two aligned 16B vector loads, each B = one aligned 32B vector load.
// Generic path covers conv0 (Cin==3).
// transpose (k=2,s=2): grid.z = parity (pz,py,px); out[2i+p] uses W[1-p]
// (jax conv_transpose, no kernel flip); renorm scale = 8 * mask_parent.
// ---------------------------------------------------------------------------
__global__ __launch_bounds__(32)
void pconv_wmma_kernel(const _Float16* __restrict__ xm, const float* __restrict__ mk_in,
                       const _Float16* __restrict__ wp,
                       float* __restrict__ out, float* __restrict__ mout,
                       int Din, int Dout, int Cin, int Cout, int Cp,
                       int ks, int stride, int pad, int transpose) {
  const int lane = threadIdx.x;
  const int g  = lane >> 4;          // half-wave
  const int ln = lane & 15;
  const int mtile = blockIdx.x, ntile = blockIdx.y, par = blockIdx.z;
  const int Vin   = Din * Din * Din;
  const int M     = transpose ? Vin : Dout * Dout * Dout;
  const int kvol  = transpose ? 1 : ks * ks * ks;
  const int Ktot  = kvol * Cin;
  const int kcube = ks * ks * ks;
  const int pz = (par >> 2) & 1, py = (par >> 1) & 1, px = par & 1;
  const int wrow_base = transpose ? ((((1 - pz) * 2 + (1 - py)) * 2 + (1 - px)) * Cin) : 0;

  const int  arow    = mtile * 16 + ln;   // this lane's A-matrix row (voxel)
  const bool arow_ok = arow < M;
  int oz = 0, oy = 0, ox = 0;
  {
    int t = arow_ok ? arow : 0;
    const int Dd = transpose ? Din : Dout;
    ox = t % Dd; t /= Dd; oy = t % Dd; oz = t / Dd;
  }

  __shared__ float s_scale[16];
  __shared__ float s_newm[16];
  if (g == 0) {
    float ms = 0.f;
    if (arow_ok) {
      if (transpose) {
        ms = mk_in[arow];   // conv_transpose(m, ones, k=2, s=2): single parent voxel
      } else {
        for (int kz = 0; kz < ks; ++kz)
          for (int ky = 0; ky < ks; ++ky)
            for (int kx = 0; kx < ks; ++kx) {
              int iz = oz * stride + kz - pad;
              int iy = oy * stride + ky - pad;
              int ix = ox * stride + kx - pad;
              if ((unsigned)iz < (unsigned)Din && (unsigned)iy < (unsigned)Din &&
                  (unsigned)ix < (unsigned)Din)
                ms += mk_in[(iz * Din + iy) * Din + ix];
            }
      }
    }
    float nm = (ms > 0.f) ? 1.f : 0.f;
    s_newm[ln]  = nm;
    s_scale[ln] = nm * ((float)kcube / fmaxf(ms, 1.f));
  }
  __syncthreads();

  const int ncol0 = ntile * 32 + ln;        // < Cp (packed B zero-padded in N)
  const int ncol1 = ncol0 + 16;
  v8f acc0 = { 0.f, 0.f, 0.f, 0.f, 0.f, 0.f, 0.f, 0.f };
  v8f acc1 = { 0.f, 0.f, 0.f, 0.f, 0.f, 0.f, 0.f, 0.f };

  if ((Cin & 31) == 0) {
    // ---------------- fast path: vectorized fragment loads -----------------
    for (int koff = 0; koff < kvol; ++koff) {
      const _Float16* arp = nullptr;    // this lane's input-voxel channel row
      if (arow_ok) {
        if (transpose) {
          arp = xm + (size_t)arow * Cin;
        } else {
          int kz = koff / (ks * ks);
          int r2 = koff - kz * ks * ks;
          int ky = r2 / ks;
          int kx = r2 - ky * ks;
          int iz = oz * stride + kz - pad;
          int iy = oy * stride + ky - pad;
          int ix = ox * stride + kx - pad;
          if ((unsigned)iz < (unsigned)Din && (unsigned)iy < (unsigned)Din &&
              (unsigned)ix < (unsigned)Din)
            arp = xm + (size_t)((iz * Din + iy) * Din + ix) * Cin;
        }
      }
      const int kb0 = wrow_base + koff * Cin;   // multiple of 32
      for (int cc = 0; cc < Cin; cc += 32) {
        v16h a;
        if (arp) {
          h8 lo = *(const h8*)(arp + cc + 16 * g);        // K = cc + 16g .. +7
          h8 hi = *(const h8*)(arp + cc + 16 * g + 8);   // K = cc + 16g + 8 .. +7
#pragma unroll
          for (int e = 0; e < 8; ++e) { a[e] = lo[e]; a[e + 8] = hi[e]; }
        } else {
#pragma unroll
          for (int e = 0; e < 16; ++e) a[e] = (_Float16)0.f;
        }
        const int t = (kb0 + cc) >> 5;
        v16h b0 = *(const v16h*)(wp + (((size_t)t * Cp + ncol0) << 5) + 16 * g);
        v16h b1 = *(const v16h*)(wp + (((size_t)t * Cp + ncol1) << 5) + 16 * g);
        acc0 = __builtin_amdgcn_wmma_f32_16x16x32_f16(false, a, false, b0, (short)0, acc0,
                                                      false, false);
        acc1 = __builtin_amdgcn_wmma_f32_16x16x32_f16(false, a, false, b1, (short)0, acc1,
                                                      false, false);
      }
    }
  } else {
    // ---------------- generic path (conv0: Cin==3) -------------------------
    for (int kk0 = 0; kk0 < Ktot; kk0 += 32) {
      v16h a;
#pragma unroll
      for (int e = 0; e < 16; ++e) {
        int kka = kk0 + e + 16 * g;
        _Float16 av = (_Float16)0.f;
        if (arow_ok && kka < Ktot) {
          int koff = kka / Cin;
          int ci   = kka - koff * Cin;
          int kz = koff / (ks * ks);
          int r2 = koff - kz * ks * ks;
          int ky = r2 / ks;
          int kx = r2 - ky * ks;
          int iz = oz * stride + kz - pad;
          int iy = oy * stride + ky - pad;
          int ix = ox * stride + kx - pad;
          if ((unsigned)iz < (unsigned)Din && (unsigned)iy < (unsigned)Din &&
              (unsigned)ix < (unsigned)Din)
            av = xm[(size_t)((iz * Din + iy) * Din + ix) * Cin + ci];
        }
        a[e] = av;
      }
      const int t = kk0 >> 5;
      v16h b0 = *(const v16h*)(wp + (((size_t)t * Cp + ncol0) << 5) + 16 * g);
      v16h b1 = *(const v16h*)(wp + (((size_t)t * Cp + ncol1) << 5) + 16 * g);
      acc0 = __builtin_amdgcn_wmma_f32_16x16x32_f16(false, a, false, b0, (short)0, acc0,
                                                    false, false);
      acc1 = __builtin_amdgcn_wmma_f32_16x16x32_f16(false, a, false, b1, (short)0, acc1,
                                                    false, false);
    }
  }

  // Epilogue: D layout lane l -> N=l%16, VGPR r -> M=r+8*g
#pragma unroll
  for (int r = 0; r < 8; ++r) {
    int mrow = mtile * 16 + r + 8 * g;
    if (mrow < M) {
      int vout;
      if (transpose) {
        int t = mrow; int ix = t % Din; t /= Din; int iy = t % Din; int iz = t / Din;
        vout = ((2 * iz + pz) * Dout + (2 * iy + py)) * Dout + (2 * ix + px);
      } else {
        vout = mrow;
      }
      float sc = s_scale[r + 8 * g];
      if (ncol0 < Cout) out[(size_t)vout * Cout + ncol0] = acc0[r] * sc;
      if (ncol1 < Cout) out[(size_t)vout * Cout + ncol1] = acc1[r] * sc;
    }
  }
  if (ntile == 0 && g == 0 && arow_ok) {
    int vout;
    if (transpose) {
      int t = arow; int ix = t % Din; t /= Din; int iy = t % Din; int iz = t / Din;
      vout = ((2 * iz + pz) * Dout + (2 * iy + py)) * Dout + (2 * ix + px);
    } else {
      vout = arow;
    }
    mout[vout] = s_newm[ln];
  }
}

// ---------------------------------------------------------------------------
// Host orchestration
// ---------------------------------------------------------------------------
extern "C" void kernel_launch(void* const* d_in, const int* in_sizes, int n_in,
                              void* d_out, int out_size, void* d_ws, size_t ws_size,
                              hipStream_t stream) {
  (void)in_sizes; (void)n_in; (void)out_size; (void)ws_size;

  const float* x_in  = (const float*)d_in[0];   // [1,3,32,32,32] NCDHW
  const float* mask0 = (const float*)d_in[1];   // [1,1,32,32,32]
  int ip = 2;
  auto W = [&]() -> const float* { return (const float*)d_in[ip++]; };

  char*  base = (char*)d_ws;
  size_t off  = 0;
  auto alloc  = [&](size_t bytes) -> void* {
    size_t a = (off + 255) & ~(size_t)255;
    off = a + bytes;
    return (void*)(base + a);
  };
  auto falloc = [&](size_t n) -> float*    { return (float*)alloc(n * sizeof(float)); };
  auto halloc = [&](size_t n) -> _Float16* { return (_Float16*)alloc(n * sizeof(_Float16)); };
  auto g256   = [](int n) { return dim3((unsigned)((n + 255) / 256)); };

  auto inorm_h = [&](const float* xin, const float* mm, const float* g, const float* b,
                     const float* res, float* outp, int V_, int C_, int mode) {
    inorm_kernel<<<dim3((unsigned)C_), dim3(256), 0, stream>>>(xin, mm, g, b, res, outp,
                                                               V_, C_, mode);
  };

  auto pconv_h = [&](const float* xin, const float* min_, const float* wf,
                     int Din_, int Cin_, int Cout_, int ks_, int stride_, int tr_,
                     float* outp, float* moutp, int chfirst) {
    int Vin_  = Din_ * Din_ * Din_;
    int Dout_ = tr_ ? Din_ * 2 : (stride_ == 2 ? Din_ / 2 : Din_);
    int pad_  = (stride_ == 1) ? ks_ / 2 : 0;
    int Cp    = ((Cout_ + 31) / 32) * 32;       // 32-wide N tiles per wave
    int Ktot  = ks_ * ks_ * ks_ * Cin_;
    int Ktiles = (Ktot + 31) / 32;
    size_t mk = off;
    _Float16* xm16 = halloc((size_t)Vin_ * Cin_);
    _Float16* wp   = halloc((size_t)Ktiles * Cp * 32);
    int nxm = Vin_ * Cin_;
    premul_kernel<<<g256(nxm), dim3(256), 0, stream>>>(xin, min_, xm16, Vin_, Cin_, chfirst);
    int npk = Ktiles * Cp * 32;
    packB_kernel<<<g256(npk), dim3(256), 0, stream>>>(wf, wp, Ktot, Cout_, Cp);
    int M = tr_ ? Vin_ : Dout_ * Dout_ * Dout_;
    dim3 grid((unsigned)((M + 15) / 16), (unsigned)(Cp / 32), tr_ ? 8u : 1u);
    pconv_wmma_kernel<<<grid, dim3(32), 0, stream>>>(
        (const _Float16*)xm16, min_, (const _Float16*)wp, outp, moutp,
        Din_, Dout_, Cin_, Cout_, Cp, ks_, stride_, pad_, tr_);
    off = mk;  // release temps (safe: single-stream ordering)
  };

  // BasicBlock: conv3-IN-lrelu-conv3-IN (+1x1 ds-IN) -> lrelu(out+res)
  auto run_block_h = [&](const float* xin, const float* min_, int D_, int Cin_, int Cout_,
                         float* outp, float* moutp) {
    const float* c1 = W(); const float* n1g = W(); const float* n1b = W();
    const float* c2 = W(); const float* n2g = W(); const float* n2b = W();
    const float* ds = nullptr; const float* dsg = nullptr; const float* dsb = nullptr;
    if (Cin_ != Cout_) { ds = W(); dsg = W(); dsb = W(); }
    int V_ = D_ * D_ * D_;
    size_t mk = off;
    float* t1 = falloc((size_t)V_ * Cout_); float* m1 = falloc(V_);
    pconv_h(xin, min_, c1, D_, Cin_, Cout_, 3, 1, 0, t1, m1, 0);
    inorm_h(t1, m1, n1g, n1b, nullptr, t1, V_, Cout_, 1);   // in-place IN+lrelu
    float* t2 = falloc((size_t)V_ * Cout_);
    pconv_h(t1, m1, c2, D_, Cout_, Cout_, 3, 1, 0, t2, moutp, 0);
    const float* res = xin;
    if (ds) {
      float* r = falloc((size_t)V_ * Cout_); float* mtmp = falloc(V_);
      pconv_h(xin, min_, ds, D_, Cin_, Cout_, 1, 1, 0, r, mtmp, 0);
      inorm_h(r, min_, dsg, dsb, nullptr, r, V_, Cout_, 0);  // in-place IN (mask = input mask)
      res = r;
    }
    inorm_h(t2, moutp, n2g, n2b, res, outp, V_, Cout_, 2);   // IN + res + lrelu
    off = mk;
  };

  auto enc_h = [&](const float* xin, const float* min_, int Din_, int Cin_, int Cout_,
                   float* outp, float* moutp) {
    const float* w = W(); const float* g = W(); const float* b = W();
    int Do_ = Din_ / 2; int Vo = Do_ * Do_ * Do_;
    size_t mk = off;
    float* o = falloc((size_t)Vo * Cout_);
    pconv_h(xin, min_, w, Din_, Cin_, Cout_, 2, 2, 0, o, moutp, 0);
    inorm_h(o, moutp, g, b, nullptr, outp, Vo, Cout_, 1);
    off = mk;
  };

  auto dec_h = [&](const float* xin, const float* min_, int Din_, int Cin_, int Ctr,
                   const float* skip, const float* skipm, int Cskip, int Cb,
                   float* outp, float* moutp) {
    const float* trw = W(); const float* tg = W(); const float* tb = W();
    int Do_ = Din_ * 2; int Vo = Do_ * Do_ * Do_;
    size_t mk = off;
    float* o = falloc((size_t)Vo * Ctr); float* mo = falloc(Vo);
    pconv_h(xin, min_, trw, Din_, Cin_, Ctr, 2, 2, 1, o, mo, 0);
    inorm_h(o, mo, tg, tb, nullptr, o, Vo, Ctr, 1);          // in-place
    int Ccat = Ctr + Cskip;
    float* cat = falloc((size_t)Vo * Ccat);
    concat_kernel<<<g256(Vo * Ccat), dim3(256), 0, stream>>>(o, skip, cat, Vo, Ctr, Cskip);
    float* mcat = falloc(Vo);
    maskclip_kernel<<<g256(Vo), dim3(256), 0, stream>>>(mo, skipm, mcat, Vo);
    float* bo = falloc((size_t)Vo * Cb); float* bm = falloc(Vo);
    run_block_h(cat, mcat, Do_, Ccat, Cb, bo, bm);
    run_block_h(bo, bm, Do_, Cb, Cb, outp, moutp);
    off = mk;
  };

  const int D0 = 32, D1 = 16, D2 = 8, D3 = 4, D4 = 2;
  const int V0 = D0 * D0 * D0, V1 = D1 * D1 * D1, V2 = D2 * D2 * D2;
  const int V3 = D3 * D3 * D3, V4 = D4 * D4 * D4;

  // ---- stem: conv0 (k=5, SAME) + IN + lrelu ------------------------------
  float* out_p1 = falloc((size_t)V0 * 32); float* m_p1 = falloc(V0);
  {
    const float* c0 = W(); const float* g0 = W(); const float* b0 = W();
    size_t mk = off;
    float* o0 = falloc((size_t)V0 * 32);
    pconv_h(x_in, mask0, c0, D0, 3, 32, 5, 1, 0, o0, m_p1, /*chfirst=*/1);
    inorm_h(o0, m_p1, g0, b0, nullptr, out_p1, V0, 32, 1);
    off = mk;
  }

  // ---- encoder -----------------------------------------------------------
  float* b1 = falloc((size_t)V1 * 32); float* mb1 = falloc(V1);
  {
    size_t mk = off;
    float* x1 = falloc((size_t)V1 * 32); float* mx1 = falloc(V1);
    enc_h(out_p1, m_p1, D0, 32, 32, x1, mx1);                      // conv1/bn1
    float* t = falloc((size_t)V1 * 32); float* mt = falloc(V1);
    run_block_h(x1, mx1, D1, 32, 32, t, mt);                       // b1_0
    run_block_h(t, mt, D1, 32, 32, b1, mb1);                       // b1_1
    off = mk;
  }
  float* b2 = falloc((size_t)V2 * 64); float* mb2 = falloc(V2);
  {
    size_t mk = off;
    float* x2 = falloc((size_t)V2 * 32); float* mx2 = falloc(V2);
    enc_h(b1, mb1, D1, 32, 32, x2, mx2);                           // conv2/bn2
    float* t = falloc((size_t)V2 * 64); float* mt = falloc(V2);
    run_block_h(x2, mx2, D2, 32, 64, t, mt);                       // b2_0 (ds)
    run_block_h(t, mt, D2, 64, 64, b2, mb2);                       // b2_1
    off = mk;
  }
  float* b3 = falloc((size_t)V3 * 128); float* mb3 = falloc(V3);
  {
    size_t mk = off;
    float* x3 = falloc((size_t)V3 * 64); float* mx3 = falloc(V3);
    enc_h(b2, mb2, D2, 64, 64, x3, mx3);                           // conv3/bn3
    float* t = falloc((size_t)V3 * 128); float* mt = falloc(V3);
    run_block_h(x3, mx3, D3, 64, 128, t, mt);                      // b3_0 (ds)
    run_block_h(t, mt, D3, 128, 128, b3, mb3);                     // b3_1
    off = mk;
  }
  float* e4 = falloc((size_t)V4 * 256); float* me4 = falloc(V4);
  {
    size_t mk = off;
    float* x4 = falloc((size_t)V4 * 128); float* mx4 = falloc(V4);
    enc_h(b3, mb3, D3, 128, 128, x4, mx4);                         // conv4/bn4
    float* t = falloc((size_t)V4 * 256); float* mt = falloc(V4);
    run_block_h(x4, mx4, D4, 128, 256, t, mt);                     // b4_0 (ds)
    run_block_h(t, mt, D4, 256, 256, e4, me4);                     // b4_1
    off = mk;
  }

  // ---- decoder -----------------------------------------------------------
  float* d3 = falloc((size_t)V3 * 256); float* md3 = falloc(V3);
  dec_h(e4, me4, D4, 256, 256, b3, mb3, 128, 256, d3, md3);        // tr4, b5_*
  float* d2 = falloc((size_t)V2 * 128); float* md2 = falloc(V2);
  dec_h(d3, md3, D3, 256, 128, b2, mb2, 64, 128, d2, md2);         // tr5, b6_*
  float* d1 = falloc((size_t)V1 * 96); float* md1 = falloc(V1);
  dec_h(d2, md2, D2, 128, 96, b1, mb1, 32, 96, d1, md1);           // tr6, b7_*
  float* d0 = falloc((size_t)V0 * 96); float* md0 = falloc(V0);
  dec_h(d1, md1, D1, 96, 96, out_p1, m_p1, 32, 96, d0, md0);       // tr7, b8_*

  // ---- final 1x1 conv + bias, write NCDHW y then mask --------------------
  {
    const float* fw = W(); const float* fb = W();
    size_t mk = off;
    float* yf = falloc((size_t)V0 * 20); float* mf = falloc(V0);
    pconv_h(d0, md0, fw, D0, 96, 20, 1, 1, 0, yf, mf, 0);
    final_kernel<<<g256(V0 * 21), dim3(256), 0, stream>>>(yf, fb, mf, (float*)d_out, V0, 20);
    off = mk;
  }
}
